// DeepSeekBlock_74019466379283
// MI455X (gfx1250) — matrix operand order repaired and checked by reading, not compiled, not run on hardware
//
#include <hip/hip_runtime.h>
#include <hip/hip_bf16.h>
#include <stdint.h>

// ---------------------------------------------------------------------------
// DeepSeek block (MLA + MoE) for gfx1250 (MI455X), bf16 WMMA + fp32 accumulate.
// Wave32; 16x16x32 bf16 WMMA (ISA 7.12); TDM tile DMA (ISA ch8) for GEMM
// staging; async global->LDS loads for attention staging.
// ---------------------------------------------------------------------------

typedef __attribute__((ext_vector_type(16))) __bf16 v16bf;
typedef __attribute__((ext_vector_type(8)))  float  v8f;
typedef __attribute__((ext_vector_type(4)))  unsigned int u32x4;
typedef __attribute__((ext_vector_type(8)))  unsigned int u32x8;
typedef int v4i_gcc __attribute__((__vector_size__(16)));   // matches async builtin param

#define AS1 __attribute__((address_space(1)))
#define AS3 __attribute__((address_space(3)))

#if defined(__AMDGCN__)
#define USE_TDM 1
#else
#define USE_TDM 0
#endif

#if defined(__AMDGCN__) && __has_builtin(__builtin_amdgcn_global_load_async_to_lds_b128)
#define USE_ASYNC_LDS 1
#else
#define USE_ASYNC_LDS 0
#endif

union FragB16 { unsigned u[8]; v16bf v; };

__device__ __forceinline__ unsigned short f2bf(float f) {
  unsigned u = __float_as_uint(f);
  u = u + 0x7FFFu + ((u >> 16) & 1u);   // round-to-nearest-even
  return (unsigned short)(u >> 16);
}

__device__ __forceinline__ v8f zero8() {
  v8f z;
#pragma unroll
  for (int i = 0; i < 8; ++i) z[i] = 0.0f;
  return z;
}

#if USE_TDM
// Tensor Data Mover: DMA one 2-D bf16 tile (tileW x tileH) from global into
// LDS, applying row padding in-flight (padInterval/padAmount codes per ISA
// 8.4). 2-SGPR-group descriptor form (VADDR2/3 = NULL -> tensors up to 2D).
__device__ __forceinline__ void tdm_load_tile_2d(unsigned ldsAddr,
                                                 const unsigned short* gsrc,
                                                 int rowLen, int nRows,
                                                 int tileW, int tileH,
                                                 int padInterval, int padAmount)
{
  const unsigned long long ga = (unsigned long long)(uintptr_t)gsrc;
  u32x4 g0;
  g0[0] = 1u;                                   // count=1, user descriptor
  g0[1] = ldsAddr;                              // lds_addr (bytes)
  g0[2] = (unsigned)ga;                         // global_addr[31:0]
  g0[3] = (unsigned)(ga >> 32) | (2u << 30);    // global_addr[56:32] | type=2
  u32x8 g1;
  g1[0] = (1u << 16) | (1u << 20) |             // data_size=2B, pad_enable
          ((unsigned)padInterval << 22) | ((unsigned)padAmount << 25);
  g1[1] = ((unsigned)rowLen & 0xFFFFu) << 16;                           // tensor_dim0 lo
  g1[2] = (((unsigned)rowLen >> 16) & 0xFFFFu) |
          (((unsigned)nRows & 0xFFFFu) << 16);                          // dim0 hi | dim1 lo
  g1[3] = (((unsigned)nRows >> 16) & 0xFFFFu) |
          (((unsigned)tileW & 0xFFFFu) << 16);                          // dim1 hi | tile_dim0
  g1[4] = (unsigned)tileH;                                              // tile_dim1 (tile_dim2=0)
  g1[5] = (unsigned)rowLen;                                             // tensor_dim0_stride lo32
  g1[6] = 0u;                                                           // stride0 hi | stride1 lo
  g1[7] = 0u;
  asm volatile("tensor_load_to_lds %0, %1" :: "s"(g0), "s"(g1) : "memory");
}

__device__ __forceinline__ void wait_tensorcnt0() {
#if __has_builtin(__builtin_amdgcn_s_wait_tensorcnt)
  __builtin_amdgcn_s_wait_tensorcnt(0);
#else
  asm volatile("s_wait_tensorcnt 0" ::: "memory");
#endif
}
#endif

#if USE_ASYNC_LDS
__device__ __forceinline__ void async_b128(const void* g, void* l) {
  __builtin_amdgcn_global_load_async_to_lds_b128(
      (AS1 v4i_gcc*)(uintptr_t)g, (AS3 v4i_gcc*)l, 0, 0);
}
__device__ __forceinline__ void wait_asynccnt0() {
#if __has_builtin(__builtin_amdgcn_s_wait_asynccnt)
  __builtin_amdgcn_s_wait_asynccnt(0);
#else
  asm volatile("s_wait_asynccnt 0" ::: "memory");
#endif
}
#endif

// ---------------------------------------------------------------------------
// Generic GEMM: C[M,N] = A[M,K](bf16) * B[K,N](bf16), fp32 accum.
// mode 0: C = acc            mode 1: C = acc + aux[m*ldc+n] (residual)
// mode 2: C += aux[m*auxStride] * acc (per-row scaled accumulate)
// Cb (optional): bf16 mirror of the fp32 result for chained GEMMs.
// Macro tile 128x64, 8 waves (4 along M x 2 along N), each wave 32x32.
// A/B tiles staged by the Tensor Data Mover (wave 0 issues the DMAs).
// ---------------------------------------------------------------------------
__global__ __launch_bounds__(256)
void wmma_gemm_bf16(const unsigned short* __restrict__ A,
                    const unsigned short* __restrict__ Bm,
                    float* __restrict__ C,
                    unsigned short* __restrict__ Cb,
                    int M, int N, int K, int lda, int ldb, int ldc,
                    int mode, const float* __restrict__ aux, int auxStride)
{
  __shared__ unsigned short As[128][40];   // 80B rows (32 data + 8 pad) -> 16B aligned
  __shared__ unsigned short Bs[32][72];    // 144B rows (64 data + 8 pad) -> 16B aligned

  const int tid  = threadIdx.x;
  const int lane = tid & 31;
  const int wave = tid >> 5;
  const int half = (lane >> 4) & 1;
  const int ln   = lane & 15;
  const int wm   = wave & 3;
  const int wn   = wave >> 2;
  const int mBase = blockIdx.y * 128;
  const int nBase = blockIdx.x * 64;

  v8f acc[2][2];
#pragma unroll
  for (int i = 0; i < 2; ++i)
#pragma unroll
    for (int j = 0; j < 2; ++j) acc[i][j] = zero8();

  const int aRow = tid >> 1;
  const int aCol = (tid & 1) * 16;
  const int bRow = tid >> 3;
  const int bCol = (tid & 7) * 8;

  for (int k0 = 0; k0 < K; k0 += 32) {
#if USE_TDM
    // Tile DMA via TDM: A 128x32 (pad every 16 dwords by 4 dwords -> 40-elem
    // rows), B 32x64 (pad every 32 dwords by 4 dwords -> 72-elem rows).
    if (wave == 0) {
      tdm_load_tile_2d((unsigned)(uintptr_t)&As[0][0],
                       A + (size_t)mBase * lda + k0, lda, M, 32, 128, 3, 3);
      tdm_load_tile_2d((unsigned)(uintptr_t)&Bs[0][0],
                       Bm + (size_t)k0 * ldb + nBase, ldb, K, 64, 32, 4, 3);
      wait_tensorcnt0();
    }
#else
    {
      const uint4* srcA = (const uint4*)(A + (size_t)(mBase + aRow) * lda + k0 + aCol);
      *(uint4*)(&As[aRow][aCol])     = srcA[0];
      *(uint4*)(&As[aRow][aCol + 8]) = srcA[1];
      const uint4* srcB = (const uint4*)(Bm + (size_t)(k0 + bRow) * ldb + nBase + bCol);
      *(uint4*)(&Bs[bRow][bCol]) = srcB[0];
    }
#endif
    if (k0 + 32 < K) {
      __builtin_prefetch(A  + (size_t)(mBase + aRow) * lda + (k0 + 32) + aCol, 0, 1);
      __builtin_prefetch(Bm + (size_t)(k0 + 32 + bRow) * ldb + nBase + bCol,  0, 1);
    }
    __syncthreads();

    // A fragments: 16-bit A 16x32 layout (ISA 7.12.2): lane=M, K pairs per VGPR
    FragB16 af[2], bf[2];
#pragma unroll
    for (int sm = 0; sm < 2; ++sm) {
      const int m = wm * 32 + sm * 16 + ln;
#pragma unroll
      for (int v = 0; v < 8; ++v) {
        const int kp = (v < 4) ? (v * 2 + half * 8) : (16 + (v - 4) * 2 + half * 8);
        af[sm].u[v] = *(const unsigned*)(&As[m][kp]);
      }
    }
    // B fragments: lane=N, K striped across VGPRs (same K order as A fragments)
#pragma unroll
    for (int sn = 0; sn < 2; ++sn) {
      const int n = wn * 32 + sn * 16 + ln;
#pragma unroll
      for (int v = 0; v < 8; ++v) {
        const int k = (v < 4) ? (v * 2 + half * 8) : (16 + (v - 4) * 2 + half * 8);
        unsigned lo = Bs[k][n];
        unsigned hi = Bs[k + 1][n];
        bf[sn].u[v] = lo | (hi << 16);
      }
    }
#pragma unroll
    for (int sm = 0; sm < 2; ++sm)
#pragma unroll
      for (int sn = 0; sn < 2; ++sn)
        acc[sm][sn] = __builtin_amdgcn_wmma_f32_16x16x32_bf16(
            false, af[sm].v, false, bf[sn].v, (short)0, acc[sm][sn], false, false);
    __syncthreads();
  }

  // Epilogue: C fragment layout -> VGPR r: M = r + half*8, N = ln
#pragma unroll
  for (int sm = 0; sm < 2; ++sm)
#pragma unroll
    for (int sn = 0; sn < 2; ++sn)
#pragma unroll
      for (int r = 0; r < 8; ++r) {
        const int gm = mBase + wm * 32 + sm * 16 + r + half * 8;
        const int gn = nBase + wn * 32 + sn * 16 + ln;
        const size_t idx = (size_t)gm * ldc + gn;
        float v = acc[sm][sn][r];
        if (mode == 1) {
          v += aux[idx];
          C[idx] = v;
        } else if (mode == 2) {
          C[idx] += aux[(size_t)gm * auxStride] * v;
        } else {
          C[idx] = v;
        }
        if (Cb) Cb[idx] = f2bf(v);
      }
}

// ---------------------------------------------------------------------------
// Flash attention, causal, HD=64, S=1024. Q/K/V in [B*NH][S][64] bf16.
// Block: 128 query rows, 8 waves (16 rows each). K-tiles of 64 keys staged
// with async global->LDS loads (ASYNCcnt). WMMA for QK^T and P*V.
// ---------------------------------------------------------------------------
__global__ __launch_bounds__(256)
void flash_attn_wmma(const unsigned short* __restrict__ Qh,
                     const unsigned short* __restrict__ Kh,
                     const unsigned short* __restrict__ Vh,
                     float* __restrict__ Y)
{
  __shared__ unsigned short Ks[64][72];
  __shared__ unsigned short Vs[64][72];
  __shared__ unsigned short Ps[8][16][72];   // per-wave P staging (A operand)

  const int tid  = threadIdx.x;
  const int lane = tid & 31;
  const int wave = tid >> 5;
  const int half = (lane >> 4) & 1;
  const int ln   = lane & 15;
  const int bh   = blockIdx.y;          // b*16 + h
  const int b    = bh >> 4;
  const int h    = bh & 15;
  const int q0   = blockIdx.x * 128;
  const int qBase = q0 + wave * 16;

  // Q fragments (16 rows x 64 dim per wave) held in registers all kernel.
  FragB16 qf[2];
  {
    const unsigned short* qrow = Qh + ((size_t)bh * 1024 + qBase + ln) * 64;
#pragma unroll
    for (int c = 0; c < 2; ++c)
#pragma unroll
      for (int v = 0; v < 8; ++v) {
        const int kp = (v < 4) ? (v * 2 + half * 8) : (16 + (v - 4) * 2 + half * 8);
        qf[c].u[v] = *(const unsigned*)(qrow + c * 32 + kp);
      }
  }

  v8f oacc[4];
#pragma unroll
  for (int i = 0; i < 4; ++i) oacc[i] = zero8();
  float mrow[8], lrow[8];
#pragma unroll
  for (int r = 0; r < 8; ++r) { mrow[r] = -1e30f; lrow[r] = 0.0f; }

  const int row  = tid >> 2;
  const int colb = (tid & 3) * 16;
  const int ktEnd = (q0 >> 6) + 2;       // causal: only tiles up to the diagonal

  for (int kt = 0; kt < ktEnd; ++kt) {
    {
      const unsigned short* kg = Kh + ((size_t)bh * 1024 + kt * 64 + row) * 64 + colb;
      const unsigned short* vg = Vh + ((size_t)bh * 1024 + kt * 64 + row) * 64 + colb;
#if USE_ASYNC_LDS
      async_b128(kg,     &Ks[row][colb]);
      async_b128(kg + 8, &Ks[row][colb + 8]);
      async_b128(vg,     &Vs[row][colb]);
      async_b128(vg + 8, &Vs[row][colb + 8]);
      wait_asynccnt0();
#else
      *(uint4*)(&Ks[row][colb])     = ((const uint4*)kg)[0];
      *(uint4*)(&Ks[row][colb + 8]) = ((const uint4*)kg)[1];
      *(uint4*)(&Vs[row][colb])     = ((const uint4*)vg)[0];
      *(uint4*)(&Vs[row][colb + 8]) = ((const uint4*)vg)[1];
#endif
    }
    __syncthreads();

    // S = Q K^T : B operand B[k=d][n=key] read from Ks[key][d] (contig in d)
    v8f sc[4];
#pragma unroll
    for (int sn = 0; sn < 4; ++sn) {
      sc[sn] = zero8();
#pragma unroll
      for (int c = 0; c < 2; ++c) {
        FragB16 kf;
        const int key = sn * 16 + ln;
#pragma unroll
        for (int v = 0; v < 8; ++v) {
          const int k = c * 32 + ((v < 4) ? (v * 2 + half * 8) : (16 + (v - 4) * 2 + half * 8));
          kf.u[v] = *(const unsigned*)(&Ks[key][k]);
        }
        sc[sn] = __builtin_amdgcn_wmma_f32_16x16x32_bf16(
            false, qf[c].v, false, kf.v, (short)0, sc[sn], false, false);
      }
    }

    // scale (1/sqrt(64)) + causal mask
#pragma unroll
    for (int sn = 0; sn < 4; ++sn) {
      const int kj = kt * 64 + sn * 16 + ln;
#pragma unroll
      for (int r = 0; r < 8; ++r) {
        const int qi = qBase + r + half * 8;
        const float s = sc[sn][r] * 0.125f;
        sc[sn][r] = (kj > qi) ? -1e30f : s;
      }
    }

    // Online softmax; rows live in 16-lane groups -> shfl_xor width 16.
#pragma unroll
    for (int r = 0; r < 8; ++r) {
      float mx = fmaxf(fmaxf(sc[0][r], sc[1][r]), fmaxf(sc[2][r], sc[3][r]));
#pragma unroll
      for (int off = 8; off > 0; off >>= 1)
        mx = fmaxf(mx, __shfl_xor(mx, off, 16));
      const float mnew  = fmaxf(mrow[r], mx);
      const float alpha = __expf(mrow[r] - mnew);
      mrow[r] = mnew;
      float rs = 0.0f;
#pragma unroll
      for (int sn = 0; sn < 4; ++sn) {
        const float p = __expf(sc[sn][r] - mnew);
        sc[sn][r] = p;
        rs += p;
      }
#pragma unroll
      for (int off = 8; off > 0; off >>= 1)
        rs += __shfl_xor(rs, off, 16);
      lrow[r] = lrow[r] * alpha + rs;
#pragma unroll
      for (int sn = 0; sn < 4; ++sn) oacc[sn][r] = oacc[sn][r] * alpha;
    }

    // P -> per-wave LDS (bf16) so it can be reloaded in A-fragment layout.
#pragma unroll
    for (int sn = 0; sn < 4; ++sn)
#pragma unroll
      for (int r = 0; r < 8; ++r)
        Ps[wave][r + half * 8][sn * 16 + ln] = f2bf(sc[sn][r]);
    asm volatile("s_wait_dscnt 0" ::: "memory");   // wave-local LDS RAW fence

    // O += P V
#pragma unroll
    for (int c = 0; c < 2; ++c) {
      FragB16 pf;
#pragma unroll
      for (int v = 0; v < 8; ++v) {
        const int kp = (v < 4) ? (v * 2 + half * 8) : (16 + (v - 4) * 2 + half * 8);
        pf.u[v] = *(const unsigned*)(&Ps[wave][ln][c * 32 + kp]);
      }
#pragma unroll
      for (int sn = 0; sn < 4; ++sn) {
        FragB16 vf;
#pragma unroll
        for (int v = 0; v < 8; ++v) {
          const int key = c * 32 + ((v < 4) ? (v * 2 + half * 8) : (16 + (v - 4) * 2 + half * 8));
          unsigned lo = Vs[key][sn * 16 + ln];
          unsigned hi = Vs[key + 1][sn * 16 + ln];
          vf.u[v] = lo | (hi << 16);
        }
        oacc[sn] = __builtin_amdgcn_wmma_f32_16x16x32_bf16(
            false, pf.v, false, vf.v, (short)0, oacc[sn], false, false);
      }
    }
    __syncthreads();
  }

  // Normalize and scatter to Y[t, h*64 + d] (fp32 [T,1024])
#pragma unroll
  for (int r = 0; r < 8; ++r) {
    const int qi = qBase + r + half * 8;
    const float linv = 1.0f / lrow[r];
    const size_t tbase = ((size_t)b * 1024 + qi) * 1024 + h * 64;
#pragma unroll
    for (int sn = 0; sn < 4; ++sn)
      Y[tbase + sn * 16 + ln] = oacc[sn][r] * linv;
  }
}

// ---------------------------------------------------------------------------
// Elementwise / small kernels
// ---------------------------------------------------------------------------
__global__ __launch_bounds__(256)
void rmsnorm_kernel(const float* __restrict__ X, const float* __restrict__ W,
                    unsigned short* __restrict__ OutB, float* __restrict__ OutF)
{
  __shared__ float red[256];
  const int t = blockIdx.x;
  const float* x = X + (size_t)t * 1024;
  float s = 0.0f;
  for (int i = threadIdx.x; i < 1024; i += 256) { const float v = x[i]; s += v * v; }
  red[threadIdx.x] = s;
  __syncthreads();
  for (int st = 128; st > 0; st >>= 1) {
    if (threadIdx.x < st) red[threadIdx.x] += red[threadIdx.x + st];
    __syncthreads();
  }
  const float inv = 1.0f / (sqrtf(red[0] * (1.0f / 1024.0f)) + 1e-5f);
  for (int i = threadIdx.x; i < 1024; i += 256) {
    const float v = x[i] * inv * W[i];
    OutB[(size_t)t * 1024 + i] = f2bf(v);
    if (OutF) OutF[(size_t)t * 1024 + i] = v;
  }
}

// RoPE on the rope-halves + repack q/k/v into [B*NH][S][64] bf16 head-major.
__global__ __launch_bounds__(256)
void rope_pack_kernel(const float* __restrict__ kf, const float* __restrict__ qf,
                      const float* __restrict__ krf, const float* __restrict__ qrf,
                      const float* __restrict__ vf,
                      unsigned short* __restrict__ Kh, unsigned short* __restrict__ Qh,
                      unsigned short* __restrict__ Vh)
{
  const int gid = blockIdx.x * blockDim.x + threadIdx.x;   // B*S*NH*64 threads
  const int d = gid & 63;
  const int h = (gid >> 6) & 15;
  const int s = (gid >> 10) & 1023;
  const int b = gid >> 20;
  const int t = (b << 10) + s;
  const size_t dst = ((size_t)(b * 16 + h) * 1024 + s) * 64 + d;

  float qo, ko;
  if (d < 32) {                      // non-rope half
    ko = kf[(size_t)t * 512 + h * 32 + d];
    qo = qf[(size_t)t * 512 + h * 32 + d];
  } else {                           // rope half: apply rotary embedding
    const int j = d - 32;
    const int i = j & 15;
    const float invf = __expf(-(float)i * (9.210340371976184f / 16.0f)); // 10000^(-i/16)
    const float ang = (float)s * invf;
    const float cs = __cosf(ang), sn = __sinf(ang);
    const size_t base = (size_t)t * 512 + h * 32;
    const float kx = krf[base + j], qx = qrf[base + j];
    const float kr = (j < 16) ? -krf[base + j + 16] : krf[base + j - 16];
    const float qr = (j < 16) ? -qrf[base + j + 16] : qrf[base + j - 16];
    ko = kx * cs + kr * sn;
    qo = qx * cs + qr * sn;
  }
  Kh[dst] = f2bf(ko);
  Qh[dst] = f2bf(qo);
  Vh[dst] = f2bf(vf[(size_t)t * 1024 + h * 64 + d]);
}

// Router: logits = xn @ W_router + bias, sigmoid, top-2 of 7, normalize.
__global__ __launch_bounds__(32)
void router_kernel(const float* __restrict__ Xn, const float* __restrict__ Wr,
                   const float* __restrict__ bias, float* __restrict__ wts)
{
  const int t = blockIdx.x;
  const int lane = threadIdx.x;
  float acc[7];
#pragma unroll
  for (int e = 0; e < 7; ++e) acc[e] = 0.0f;
  for (int i = lane; i < 1024; i += 32) {
    const float xv = Xn[(size_t)t * 1024 + i];
#pragma unroll
    for (int e = 0; e < 7; ++e) acc[e] += xv * Wr[i * 7 + e];
  }
#pragma unroll
  for (int e = 0; e < 7; ++e)
#pragma unroll
    for (int off = 16; off > 0; off >>= 1)
      acc[e] += __shfl_xor(acc[e], off, 32);
  if (lane == 0) {
    float p[7];
#pragma unroll
    for (int e = 0; e < 7; ++e) p[e] = 1.0f / (1.0f + __expf(-(acc[e] + bias[e])));
    int i0 = 0;
#pragma unroll
    for (int e = 1; e < 7; ++e) if (p[e] > p[i0]) i0 = e;
    int i1 = (i0 == 0) ? 1 : 0;
#pragma unroll
    for (int e = 0; e < 7; ++e) if (e != i0 && p[e] > p[i1]) i1 = e;
    const float s = p[i0] + p[i1];
#pragma unroll
    for (int e = 0; e < 7; ++e)
      wts[(size_t)t * 7 + e] = (e == i0) ? p[i0] / s : ((e == i1) ? p[i1] / s : 0.0f);
  }
}

__global__ __launch_bounds__(256)
void silu_mul_kernel(const float* __restrict__ g, const float* __restrict__ u,
                     unsigned short* __restrict__ out, int n)
{
  const int i = blockIdx.x * blockDim.x + threadIdx.x;
  if (i < n) {
    const float gv = g[i];
    out[i] = f2bf(gv / (1.0f + __expf(-gv)) * u[i]);
  }
}

__global__ __launch_bounds__(256)
void cast_f32_bf16_kernel(const float* __restrict__ src,
                          unsigned short* __restrict__ dst, int n)
{
  const int i = blockIdx.x * blockDim.x + threadIdx.x;
  if (i < n) dst[i] = f2bf(src[i]);
}

// ---------------------------------------------------------------------------
// Host orchestration
// ---------------------------------------------------------------------------
extern "C" void kernel_launch(void* const* d_in, const int* in_sizes, int n_in,
                              void* d_out, int out_size, void* d_ws, size_t ws_size,
                              hipStream_t stream)
{
  (void)in_sizes; (void)n_in; (void)out_size; (void)ws_size;

  constexpr int T = 4096, H = 1024, L = 256, I = 1024, NHHH = 512; // NH*HH = 512

  const float* x        = (const float*)d_in[0];
  const float* w_ln1    = (const float*)d_in[1];
  const float* w_ln2    = (const float*)d_in[2];
  const float* W_kvd    = (const float*)d_in[3];
  const float* W_qd     = (const float*)d_in[4];
  const float* W_ku     = (const float*)d_in[5];
  const float* W_qu     = (const float*)d_in[6];
  const float* W_vu     = (const float*)d_in[7];
  const float* W_ropeq  = (const float*)d_in[8];
  const float* W_ropek  = (const float*)d_in[9];
  const float* W_o      = (const float*)d_in[10];
  const float* Wg_s     = (const float*)d_in[11];
  const float* Wu_s     = (const float*)d_in[12];
  const float* Wd_s     = (const float*)d_in[13];
  const float* Wg_r     = (const float*)d_in[14];
  const float* Wu_r     = (const float*)d_in[15];
  const float* Wd_r     = (const float*)d_in[16];
  const float* W_router = (const float*)d_in[17];
  const float* r_bias   = (const float*)d_in[18];
  float* out = (float*)d_out;

  char* ws = (char*)d_ws;
  size_t off = 0;
  auto alloc = [&](size_t bytes) -> char* {
    char* p = ws + off;
    off += (bytes + 255) & ~(size_t)255;
    return p;
  };

  // bf16 weight mirrors
  unsigned short* Wb_kvd = (unsigned short*)alloc((size_t)2 * H * L);
  unsigned short* Wb_qd  = (unsigned short*)alloc((size_t)2 * H * L);
  unsigned short* Wb_ku  = (unsigned short*)alloc((size_t)2 * L * NHHH);
  unsigned short* Wb_qu  = (unsigned short*)alloc((size_t)2 * L * NHHH);
  unsigned short* Wb_vu  = (unsigned short*)alloc((size_t)2 * L * H);
  unsigned short* Wb_rq  = (unsigned short*)alloc((size_t)2 * L * NHHH);
  unsigned short* Wb_rk  = (unsigned short*)alloc((size_t)2 * H * NHHH);
  unsigned short* Wb_o   = (unsigned short*)alloc((size_t)2 * H * H);
  unsigned short* Wb_gs  = (unsigned short*)alloc((size_t)2 * H * I);
  unsigned short* Wb_us  = (unsigned short*)alloc((size_t)2 * H * I);
  unsigned short* Wb_ds  = (unsigned short*)alloc((size_t)2 * I * H);
  unsigned short* Wb_gr  = (unsigned short*)alloc((size_t)2 * 7 * H * I);
  unsigned short* Wb_ur  = (unsigned short*)alloc((size_t)2 * 7 * H * I);
  unsigned short* Wb_dr  = (unsigned short*)alloc((size_t)2 * 7 * I * H);

  // activations
  unsigned short* xn1_b   = (unsigned short*)alloc((size_t)2 * T * H);
  float*          kvlat_f = (float*)alloc((size_t)4 * T * L);
  unsigned short* kvlat_b = (unsigned short*)alloc((size_t)2 * T * L);
  float*          qlat_f  = (float*)alloc((size_t)4 * T * L);
  unsigned short* qlat_b  = (unsigned short*)alloc((size_t)2 * T * L);
  float*          k_f     = (float*)alloc((size_t)4 * T * NHHH);
  float*          q_f     = (float*)alloc((size_t)4 * T * NHHH);
  float*          qr_f    = (float*)alloc((size_t)4 * T * NHHH);
  float*          kr_f    = (float*)alloc((size_t)4 * T * NHHH);
  float*          v_f     = (float*)alloc((size_t)4 * T * H);
  unsigned short* Qh      = (unsigned short*)alloc((size_t)2 * T * H);
  unsigned short* Kh      = (unsigned short*)alloc((size_t)2 * T * H);
  unsigned short* Vh      = (unsigned short*)alloc((size_t)2 * T * H);
  float*          y_f     = (float*)alloc((size_t)4 * T * H);
  unsigned short* y_b     = (unsigned short*)alloc((size_t)2 * T * H);
  float*          x1_f    = (float*)alloc((size_t)4 * T * H);
  unsigned short* xn2_b   = (unsigned short*)alloc((size_t)2 * T * H);
  float*          xn2_f   = (float*)alloc((size_t)4 * T * H);
  float*          wts     = (float*)alloc((size_t)4 * T * 7);
  float*          g_f     = (float*)alloc((size_t)4 * T * I);
  float*          u_f     = (float*)alloc((size_t)4 * T * I);
  unsigned short* hmid_b  = (unsigned short*)alloc((size_t)2 * T * I);

  auto cast = [&](const float* s, unsigned short* d, int n) {
    cast_f32_bf16_kernel<<<(n + 255) / 256, 256, 0, stream>>>(s, d, n);
  };
  auto gemm = [&](const unsigned short* A, const unsigned short* Bm, float* C,
                  unsigned short* Cb, int M, int N, int K, int lda, int ldb, int ldc,
                  int mode, const float* aux, int auxStride) {
    dim3 grid(N / 64, M / 128);
    wmma_gemm_bf16<<<grid, 256, 0, stream>>>(A, Bm, C, Cb, M, N, K, lda, ldb, ldc,
                                             mode, aux, auxStride);
  };

  // ---- weight casts (one-time per launch; deterministic) ----
  cast(W_kvd,   Wb_kvd, H * L);
  cast(W_qd,    Wb_qd,  H * L);
  cast(W_ku,    Wb_ku,  L * NHHH);
  cast(W_qu,    Wb_qu,  L * NHHH);
  cast(W_vu,    Wb_vu,  L * H);
  cast(W_ropeq, Wb_rq,  L * NHHH);
  cast(W_ropek, Wb_rk,  H * NHHH);
  cast(W_o,     Wb_o,   H * H);
  cast(Wg_s,    Wb_gs,  H * I);
  cast(Wu_s,    Wb_us,  H * I);
  cast(Wd_s,    Wb_ds,  I * H);
  cast(Wg_r,    Wb_gr,  7 * H * I);
  cast(Wu_r,    Wb_ur,  7 * H * I);
  cast(Wd_r,    Wb_dr,  7 * I * H);

  // ---- MLA attention ----
  rmsnorm_kernel<<<T, 256, 0, stream>>>(x, w_ln1, xn1_b, nullptr);

  gemm(xn1_b,   Wb_kvd, kvlat_f, kvlat_b, T, L,    H, H, L,    L,    0, nullptr, 0);
  gemm(xn1_b,   Wb_qd,  qlat_f,  qlat_b,  T, L,    H, H, L,    L,    0, nullptr, 0);
  gemm(kvlat_b, Wb_ku,  k_f,  nullptr,    T, NHHH, L, L, NHHH, NHHH, 0, nullptr, 0);
  gemm(qlat_b,  Wb_qu,  q_f,  nullptr,    T, NHHH, L, L, NHHH, NHHH, 0, nullptr, 0);
  gemm(kvlat_b, Wb_vu,  v_f,  nullptr,    T, H,    L, L, H,    H,    0, nullptr, 0);
  gemm(qlat_b,  Wb_rq,  qr_f, nullptr,    T, NHHH, L, L, NHHH, NHHH, 0, nullptr, 0);
  gemm(xn1_b,   Wb_rk,  kr_f, nullptr,    T, NHHH, H, H, NHHH, NHHH, 0, nullptr, 0);

  rope_pack_kernel<<<(T * 16 * 64) / 256, 256, 0, stream>>>(
      k_f, q_f, kr_f, qr_f, v_f, Kh, Qh, Vh);

  {
    dim3 grid(1024 / 128, 4 * 16);   // (S/128, B*NH)
    flash_attn_wmma<<<grid, 256, 0, stream>>>(Qh, Kh, Vh, y_f);
  }
  cast(y_f, y_b, T * H);

  // x1 = x + attn @ W_o
  gemm(y_b, Wb_o, x1_f, nullptr, T, H, H, H, H, H, 1, x, 0);

  // ---- MoE ----
  rmsnorm_kernel<<<T, 256, 0, stream>>>(x1_f, w_ln2, xn2_b, xn2_f);
  router_kernel<<<T, 32, 0, stream>>>(xn2_f, W_router, r_bias, wts);

  // shared expert -> out = x1 + shared_down(silu(g)*u)
  gemm(xn2_b, Wb_gs, g_f, nullptr, T, I, H, H, I, I, 0, nullptr, 0);
  gemm(xn2_b, Wb_us, u_f, nullptr, T, I, H, H, I, I, 0, nullptr, 0);
  silu_mul_kernel<<<(T * I + 255) / 256, 256, 0, stream>>>(g_f, u_f, hmid_b, T * I);
  gemm(hmid_b, Wb_ds, out, nullptr, T, H, I, I, H, H, 1, x1_f, 0);

  // routed experts: out += wts[:,e] * down_e(silu(g_e)*u_e)
  for (int e = 0; e < 7; ++e) {
    const unsigned short* wg = Wb_gr + (size_t)e * H * I;
    const unsigned short* wu = Wb_ur + (size_t)e * H * I;
    const unsigned short* wd = Wb_dr + (size_t)e * I * H;
    gemm(xn2_b, wg, g_f, nullptr, T, I, H, H, I, I, 0, nullptr, 0);
    gemm(xn2_b, wu, u_f, nullptr, T, I, H, H, I, I, 0, nullptr, 0);
    silu_mul_kernel<<<(T * I + 255) / 256, 256, 0, stream>>>(g_f, u_f, hmid_b, T * I);
    gemm(hmid_b, wd, out, nullptr, T, H, I, I, H, H, 2, wts + e, 7);
  }
}
